// HoPEAttention_34961033789733
// MI455X (gfx1250) — compile-verified, matrix operand order checked by reading, not run on hardware
//
#include <hip/hip_runtime.h>
#include <hip/hip_bf16.h>

// Problem constants (match reference)
#define BB 2
#define TT 2048
#define DDIM 2048
#define HH 32
#define HD 64
#define MROWS (BB * TT) // 4096

typedef __attribute__((ext_vector_type(16))) __bf16 v16bf;
typedef __attribute__((ext_vector_type(8)))  float  v8f;

union TileU { uint4 q[2]; v16bf v; };

// Load one 16x32 bf16 operand tile in the CDNA5 WMMA 16-bit A layout:
// lane%16 = row, lanes 0-15 hold K = {0..7, 16..23}, lanes 16-31 hold K = {8..15, 24..31}.
// Matrix is row-major [rows, ldk] bf16; two b128 loads per lane.
__device__ __forceinline__ v16bf load_tile16(const __hip_bfloat16* __restrict__ base,
                                             int row0, int k0, int ldk, int lane) {
  const int hi = lane >> 4;
  const int r  = lane & 15;
  const char* p = (const char*)(base + (size_t)(row0 + r) * ldk + (k0 + hi * 8));
  TileU t;
  t.q[0] = *(const uint4*)(p);        // K = k0 + hi*8 .. +7
  t.q[1] = *(const uint4*)(p + 32);   // K = k0 + 16 + hi*8 .. +7
  return t.v;
}

// C[M,N] fp32 = A[M,K] bf16 * Bt[N,K]^T bf16.
// One wave -> 64x64 C tile (4x4 WMMA accumulators): 16 WMMAs per 16 b128 loads.
__global__ __launch_bounds__(256) void gemm_wmma_bf16(
    const __hip_bfloat16* __restrict__ A,
    const __hip_bfloat16* __restrict__ Bt,
    float* __restrict__ C, int M, int N, int K) {
  const int lane = threadIdx.x & 31;
  const int wave = threadIdx.x >> 5;
  const int tile = blockIdx.x * 8 + wave;
  const int ntn  = N >> 6;            // 64-wide tiles in N
  const int M0 = (tile / ntn) << 6;
  const int N0 = (tile % ntn) << 6;

  v8f acc[4][4];
  #pragma unroll
  for (int i = 0; i < 4; ++i)
    #pragma unroll
    for (int j = 0; j < 4; ++j) acc[i][j] = {};

  #pragma unroll 1
  for (int k0 = 0; k0 < K; k0 += 32) {
    if (k0 + 32 < K) {
      __builtin_prefetch(A  + (size_t)(M0 + (lane & 15)) * K + k0 + 32, 0, 1);
      __builtin_prefetch(Bt + (size_t)(N0 + (lane & 15)) * K + k0 + 32, 0, 1);
    }
    v16bf a[4], b[4];
    #pragma unroll
    for (int i = 0; i < 4; ++i) a[i] = load_tile16(A,  M0 + 16 * i, k0, K, lane);
    #pragma unroll
    for (int j = 0; j < 4; ++j) b[j] = load_tile16(Bt, N0 + 16 * j, k0, K, lane);
    #pragma unroll
    for (int i = 0; i < 4; ++i)
      #pragma unroll
      for (int j = 0; j < 4; ++j)
        acc[i][j] = __builtin_amdgcn_wmma_f32_16x16x32_bf16(
            false, a[i], false, b[j], (short)0, acc[i][j], false, false);
  }

  // 16x16 f32 D layout: VGPR r -> M = r + 8*(lane/16), lane%16 -> N.
  const int hi = lane >> 4, c = lane & 15;
  #pragma unroll
  for (int i = 0; i < 4; ++i)
    #pragma unroll
    for (int j = 0; j < 4; ++j)
      #pragma unroll
      for (int r = 0; r < 8; ++r) {
        const int m = 16 * i + r + hi * 8;
        C[(size_t)(M0 + m) * N + N0 + 16 * j + c] = acc[i][j][r];
      }
}

// Elementwise fp32 -> bf16 convert.
__global__ void cvt_bf16(const float* __restrict__ src, __hip_bfloat16* __restrict__ dst, int n) {
  int i = blockIdx.x * blockDim.x + threadIdx.x;
  if (i < n) dst[i] = __float2bfloat16(src[i]);
}

// WT[n][k] = bf16(W[k][n]) for 2048x2048 (coalesced writes).
__global__ void transpose_cvt(const float* __restrict__ W, __hip_bfloat16* __restrict__ WT) {
  int id  = blockIdx.x * blockDim.x + threadIdx.x;
  int row = id >> 11;        // n
  int col = id & 2047;       // k
  WT[id] = __float2bfloat16(W[(size_t)col * DDIM + row]);
}

// beta[row,h] = sigmoid(X[row,:] . Wbeta[:,h] + b_beta[h]); one thread per output.
__global__ __launch_bounds__(256) void beta_kernel(const float* __restrict__ X,
                                                   const float* __restrict__ Wbeta,
                                                   const float* __restrict__ bb,
                                                   float* __restrict__ beta) {
  const int h   = threadIdx.x & 31;
  const int row = blockIdx.x * 8 + (threadIdx.x >> 5);
  const float* xr = X + (size_t)row * DDIM;
  float acc = 0.f;
  for (int k = 0; k < DDIM; ++k) acc = fmaf(xr[k], Wbeta[(size_t)k * HH + h], acc);
  acc += bb[h];
  beta[(size_t)row * HH + h] = 1.f / (1.f + __expf(-acc));
}

// In-place L2 normalization of contiguous 64-element head rows; one wave per segment.
__global__ __launch_bounds__(256) void l2norm64(float* __restrict__ x) {
  const int lane = threadIdx.x & 31;
  const int seg  = blockIdx.x * 8 + (threadIdx.x >> 5);
  float* p = x + (size_t)seg * 64;
  float a = p[lane], b = p[lane + 32];
  float ss = a * a + b * b;
  #pragma unroll
  for (int off = 16; off; off >>= 1) ss += __shfl_xor(ss, off, 32);
  const float r = rsqrtf(ss + 1e-6f);
  p[lane]      = a * r;
  p[lane + 32] = b * r;
}

// Sequential Householder delta-rule scan. One 64-thread WG per (b,h); thread j
// owns column j of the 64x64 state S in registers. Double-buffered LDS broadcast
// (one barrier per step) + register prefetch of step t+1 to hide load latency.
__global__ __launch_bounds__(64) void hope_scan(const float* __restrict__ qn,
                                                const float* __restrict__ kn,
                                                const float* __restrict__ v,
                                                const float* __restrict__ beta,
                                                __hip_bfloat16* __restrict__ out) {
  const int b = blockIdx.x / HH;
  const int h = blockIdx.x % HH;
  const int j = threadIdx.x;
  __shared__ float lk[2][HD];
  __shared__ float lq[2][HD];
  float S[HD];
  #pragma unroll
  for (int i = 0; i < HD; ++i) S[i] = 0.f;

  const size_t row0 = (size_t)b * TT;
  size_t base = row0 * DDIM + (size_t)h * HD;
  float qv  = qn[base + j];
  float kv  = kn[base + j];
  float vv  = v[base + j];
  float bet = beta[row0 * HH + h];

  for (int t = 0; t < TT; ++t) {
    const int buf = t & 1;
    lq[buf][j] = qv;
    lk[buf][j] = kv;
    const float vcur = vv, bcur = bet;
    const size_t obase = base;
    __syncthreads();

    // Prefetch next step's operands while this step computes.
    if (t + 1 < TT) {
      const size_t row = row0 + t + 1;
      base = row * DDIM + (size_t)h * HD;
      qv  = qn[base + j];
      kv  = kn[base + j];
      vv  = v[base + j];
      bet = beta[row * HH + h];
    }

    float pred = 0.f;
    #pragma unroll
    for (int i = 0; i < HD; ++i) pred = fmaf(lk[buf][i], S[i], pred);
    const float del = (vcur - pred) * bcur;
    float o = 0.f;
    #pragma unroll
    for (int i = 0; i < HD; ++i) {
      S[i] = fmaf(lk[buf][i], del, S[i]);
      o    = fmaf(lq[buf][i], S[i], o);
    }
    out[obase + j] = __float2bfloat16(o);
  }
}

extern "C" void kernel_launch(void* const* d_in, const int* in_sizes, int n_in,
                              void* d_out, int out_size, void* d_ws, size_t ws_size,
                              hipStream_t stream) {
  const float* X     = (const float*)d_in[0];
  const float* Wq    = (const float*)d_in[1];
  const float* Wk    = (const float*)d_in[2];
  const float* Wv    = (const float*)d_in[3];
  const float* Wbeta = (const float*)d_in[4];
  const float* bbeta = (const float*)d_in[5];
  const float* Wo    = (const float*)d_in[6];

  char* ws = (char*)d_ws;
  size_t off = 0;
  __hip_bfloat16* Xb    = (__hip_bfloat16*)(ws + off); off += (size_t)MROWS * DDIM * 2;
  __hip_bfloat16* WqT   = (__hip_bfloat16*)(ws + off); off += (size_t)DDIM * DDIM * 2;
  __hip_bfloat16* WkT   = (__hip_bfloat16*)(ws + off); off += (size_t)DDIM * DDIM * 2;
  __hip_bfloat16* WvT   = (__hip_bfloat16*)(ws + off); off += (size_t)DDIM * DDIM * 2;
  __hip_bfloat16* WoT   = (__hip_bfloat16*)(ws + off); off += (size_t)DDIM * DDIM * 2;
  float*          qf    = (float*)(ws + off);          off += (size_t)MROWS * DDIM * 4;
  float*          kf    = (float*)(ws + off);          off += (size_t)MROWS * DDIM * 4;
  float*          vf    = (float*)(ws + off);          off += (size_t)MROWS * DDIM * 4;
  float*          betaf = (float*)(ws + off);          off += (size_t)MROWS * HH * 4;
  __hip_bfloat16* attnb = (__hip_bfloat16*)(ws + off); off += (size_t)MROWS * DDIM * 2;
  (void)ws_size; (void)in_sizes; (void)n_in; (void)out_size;

  // 1. Precision conversion (bf16 operands for WMMA), weights transposed to [N,K].
  cvt_bf16<<<(MROWS * DDIM) / 256, 256, 0, stream>>>(X, Xb, MROWS * DDIM);
  transpose_cvt<<<(DDIM * DDIM) / 256, 256, 0, stream>>>(Wq, WqT);
  transpose_cvt<<<(DDIM * DDIM) / 256, 256, 0, stream>>>(Wk, WkT);
  transpose_cvt<<<(DDIM * DDIM) / 256, 256, 0, stream>>>(Wv, WvT);
  transpose_cvt<<<(DDIM * DDIM) / 256, 256, 0, stream>>>(Wo, WoT);

  // 2. q/k/v projections via WMMA (fp32 accumulation). 64x64 tile per wave.
  const int gemm_blocks = (MROWS / 64) * (DDIM / 64) / 8; // 256
  gemm_wmma_bf16<<<gemm_blocks, 256, 0, stream>>>(Xb, WqT, qf, MROWS, DDIM, DDIM);
  gemm_wmma_bf16<<<gemm_blocks, 256, 0, stream>>>(Xb, WkT, kf, MROWS, DDIM, DDIM);
  gemm_wmma_bf16<<<gemm_blocks, 256, 0, stream>>>(Xb, WvT, vf, MROWS, DDIM, DDIM);

  // 3. beta = sigmoid(X @ Wbeta + b) in fp32.
  beta_kernel<<<MROWS / 8, 256, 0, stream>>>(X, Wbeta, bbeta, betaf);

  // 4. L2-normalize q and k per head row (in place).
  l2norm64<<<(MROWS * HH) / 8, 256, 0, stream>>>(qf);
  l2norm64<<<(MROWS * HH) / 8, 256, 0, stream>>>(kf);

  // 5. Sequential delta-rule scan, parallel over B*H = 64 heads.
  hope_scan<<<BB * HH, 64, 0, stream>>>(qf, kf, vf, betaf, attnb);

  // 6. Output projection via WMMA directly into d_out (fp32).
  gemm_wmma_bf16<<<gemm_blocks, 256, 0, stream>>>(attnb, WoT, (float*)d_out, MROWS, DDIM, DDIM);
}
